// EnsembleRSSM_45896020525407
// MI455X (gfx1250) — matrix operand order repaired and checked by reading, not compiled, not run on hardware
//
#include <hip/hip_runtime.h>

// ---------------------------------------------------------------------------
// EnsembleRSSM scan for gfx1250 (MI455X).
// 4 persistent workgroups x 512 threads (16 wave32). Each WG owns 16 batch
// rows and runs the whole T=256 recurrence. GEMMs: v_wmma_f32_16x16x32_bf16.
// Weights converted once per call into WMMA-register-layout bf16 tiles (d_ws).
// ---------------------------------------------------------------------------

typedef __bf16 v16bf __attribute__((ext_vector_type(16)));
typedef __bf16 v8bf  __attribute__((ext_vector_type(8)));
typedef float  v8f   __attribute__((ext_vector_type(8)));

// Problem constants
#define BB      64
#define TT      256
#define EE      1536
#define HID     200
#define DET     200
#define SD      1024
#define INDIM   1074

#define NTHREADS 512
#define NWAVES   16

// Padded K / tile counts (K pad -> mult of 32, N pad -> mult of 16)
#define KT_IN   34   // 1088
#define NT_IN   13   // 208
#define KT_GRU  13   // 416
#define NT_GRU  38   // 608
#define KT_H    7    // 224
#define NT_HID  13   // 208
#define NT_SD   64   // 1024
#define KT_O1   55   // 1760

#define TILE_ELEMS 512  // 32 lanes * 16 bf16

// d_ws element offsets (bf16 elements)
#define OFF_IN   ((size_t)0)
#define OFF_GRU  (OFF_IN  + (size_t)KT_IN  * NT_IN  * TILE_ELEMS)
#define OFF_1E   (OFF_GRU + (size_t)KT_GRU * NT_GRU * TILE_ELEMS)
#define SZ_1E    ((size_t)KT_H * NT_HID * TILE_ELEMS)
#define OFF_2E   (OFF_1E  + 5 * SZ_1E)
#define SZ_2E    ((size_t)KT_H * NT_SD * TILE_ELEMS)
#define OFF_O1   (OFF_2E  + 5 * SZ_2E)
#define OFF_O2   (OFF_O1  + (size_t)KT_O1 * NT_HID * TILE_ELEMS)
#define WS_ELEMS (OFF_O2  + (size_t)KT_H  * NT_SD  * TILE_ELEMS)

// LDS layout (bytes)
#define L_XIN   0                       // bf16 [16][1088]  stoch|a50|pad
#define L_GRUIN (L_XIN   + 34816)       // bf16 [16][416]   x|deter|pad
#define L_PIN   (L_GRUIN + 13312)       // bf16 [16][1760]  deter|embed|pad
#define L_HID   (L_PIN   + 56320)       // bf16 [16][224]
#define L_DBF   (L_HID   + 7168)        // bf16 [16][224]   deter bf16
#define L_DET   (L_DBF   + 7168)        // f32  [16][200]
#define L_FBUF  (L_DET   + 12800)       // f32  [16][1024]  GEMM staging
#define L_MU    (L_FBUF  + 65536)       // f32  [16]
#define L_RS    (L_MU    + 64)          // f32  [16]
#define L_ISF   (L_RS    + 64)          // u32  [16]
#define SMEM_BYTES (L_ISF + 64)

// ---------------------------------------------------------------------------
// Weight conversion: row-major f32 [K][N] -> WMMA-layout bf16 tiles.
// Tile (nt,kt) at ((nt*numKt)+kt)*512; element (lane,i):
//   k = kt*32 + (lane>>4)*16 + i ; n = nt*16 + (lane&15)   (zero padded)
// ---------------------------------------------------------------------------
__global__ void rssm_convert_tiles(const float* __restrict__ src,
                                   __bf16* __restrict__ dst,
                                   int K, int N, int numKt, int numNt) {
  int idx = blockIdx.x * blockDim.x + threadIdx.x;
  int total = numKt * numNt * TILE_ELEMS;
  if (idx >= total) return;
  int e    = idx & (TILE_ELEMS - 1);
  int tile = idx >> 9;
  int kt   = tile % numKt;
  int nt   = tile / numKt;
  int lane = e >> 4;
  int i    = e & 15;
  int k = kt * 32 + ((lane >> 4) & 1) * 16 + i;
  int n = nt * 16 + (lane & 15);
  float v = (k < K && n < N) ? src[(size_t)k * N + n] : 0.f;
  dst[idx] = (__bf16)v;
}

// ---------------------------------------------------------------------------
// 16xN = (16xKp bf16 in LDS) @ (Kp x N tiled bf16 in global), f32 out to LDS.
// All 16 waves participate; wave w handles N-tiles w, w+16, ...
// A layout (matches B tiles): lane m=lane&15; contiguous [k0+hi*16, +16).
// ---------------------------------------------------------------------------
__device__ __forceinline__ void mm_wmma(const __bf16* __restrict__ act, int astride,
                                        const __bf16* __restrict__ wt,
                                        int numKt, int numNt,
                                        float* __restrict__ out, int ostride) {
  const int lane = threadIdx.x & 31;
  const int wave = threadIdx.x >> 5;
  const __bf16* row = act + (size_t)(lane & 15) * astride;
  const int hi8 = ((lane >> 4) & 1) * 8;
  for (int nt = wave; nt < numNt; nt += NWAVES) {
    const __bf16* wtn = wt + (size_t)nt * numKt * TILE_ELEMS + (size_t)lane * 16;
    v8f acc = {0.f, 0.f, 0.f, 0.f, 0.f, 0.f, 0.f, 0.f};
    for (int kt = 0; kt < numKt; ++kt) {
      const int k0 = kt * 32;
      v8bf c0 = *(const v8bf*)(row + k0 + 2 * hi8);
      v8bf c1 = *(const v8bf*)(row + k0 + 2 * hi8 + 8);
      v16bf a = __builtin_shufflevector(c0, c1, 0, 1, 2, 3, 4, 5, 6, 7,
                                        8, 9, 10, 11, 12, 13, 14, 15);
      v16bf b = *(const v16bf*)(wtn + (size_t)kt * TILE_ELEMS);
      acc = __builtin_amdgcn_wmma_f32_16x16x32_bf16(
          false, a, false, b, (short)0, acc, false, false);
    }
    const int n = nt * 16 + (lane & 15);
    const int mb = (lane >> 4) * 8;
    #pragma unroll
    for (int r = 0; r < 8; ++r)
      out[(size_t)(mb + r) * ostride + n] = acc[r];
  }
}

// One wave per row: lanes stride the row, butterfly-reduce sum / sumsq.
__device__ __forceinline__ void ln_stats_wave(const float* __restrict__ row,
                                              const float* __restrict__ bias,
                                              int n, float* mu, float* rs) {
  const int lane = threadIdx.x & 31;
  float s = 0.f, s2 = 0.f;
  for (int j = lane; j < n; j += 32) {
    float v = row[j] + bias[j];
    s += v; s2 += v * v;
  }
  #pragma unroll
  for (int off = 16; off > 0; off >>= 1) {
    s  += __shfl_xor(s,  off, 32);
    s2 += __shfl_xor(s2, off, 32);
  }
  if (lane == 0) {
    float m = s / n;
    *mu = m;
    *rs = rsqrtf(s2 / n - m * m + 1e-5f);
  }
}

struct RssmParams {
  const float* embed;
  const float* action;
  const unsigned char* is_first;
  const int* ens_idx;
  const float *b_in, *g_in, *be_in;
  const float *b_gru, *g_gru, *be_gru;
  const float *b1e, *g1e, *be1e, *b2e;
  const float *bo1, *go1, *beo1, *bo2;
  const __bf16* ws;
  float *out_ps, *out_pq, *out_det, *out_prs, *out_prl;
};

__global__ __launch_bounds__(NTHREADS) void rssm_scan_kernel(RssmParams p) {
  extern __shared__ char smem[];
  __bf16* x_in   = (__bf16*)(smem + L_XIN);    // [16][1088]
  __bf16* gru_in = (__bf16*)(smem + L_GRUIN);  // [16][416]
  __bf16* pin    = (__bf16*)(smem + L_PIN);    // [16][1760]
  __bf16* hid    = (__bf16*)(smem + L_HID);    // [16][224]
  __bf16* dbf    = (__bf16*)(smem + L_DBF);    // [16][224]
  float*  det    = (float*)(smem + L_DET);     // [16][200]
  float*  fbuf   = (float*)(smem + L_FBUF);    // [16][1024]
  float*  mu     = (float*)(smem + L_MU);
  float*  rs     = (float*)(smem + L_RS);
  unsigned* isfl = (unsigned*)(smem + L_ISF);  // [16]

  const int tid  = threadIdx.x;
  const int wave = tid >> 5;
  const int b0   = blockIdx.x * 16;

  // ---- init persistent state & zero pad regions (pads stay zero forever) ----
  for (int i = tid; i < 16 * 1088; i += NTHREADS) x_in[i] = (__bf16)0.f;
  for (int i = tid; i < 16 * 200;  i += NTHREADS) det[i] = 0.f;
  for (int i = tid; i < 16 * 16;   i += NTHREADS) gru_in[(i >> 4) * 416 + 400 + (i & 15)] = (__bf16)0.f;
  for (int i = tid; i < 16 * 24;   i += NTHREADS) pin[(i / 24) * 1760 + 1736 + (i % 24)] = (__bf16)0.f;
  for (int i = tid; i < 16 * 24;   i += NTHREADS) {
    hid[(i / 24) * 224 + 200 + (i % 24)] = (__bf16)0.f;
    dbf[(i / 24) * 224 + 200 + (i % 24)] = (__bf16)0.f;
  }
  __syncthreads();

  for (int t = 0; t < TT; ++t) {
    // ---- phase 0: hoist per-row is_first flags ----
    if (tid < 16)
      isfl[tid] = (unsigned)p.is_first[(size_t)(b0 + tid) * TT + t];
    __syncthreads();

    // ---- phase 1: episode-start masking, action fill, embed -> bf16 ----
    for (int i = tid; i < 16 * 1024; i += NTHREADS) {
      int r = i >> 10;
      if (isfl[r]) x_in[r * 1088 + (i & 1023)] = (__bf16)0.f;
    }
    if (tid < 16) {
      int r = tid;
      bool isf = isfl[r] != 0;
      if (isf) for (int j = 0; j < DET; ++j) det[r * 200 + j] = 0.f;
      const float* ar = p.action + ((size_t)(b0 + r) * TT + t) * 6;
      for (int j = 0; j < 6; ++j)
        x_in[r * 1088 + 1024 + j] = (__bf16)(isf ? 0.f : ar[j]);
      // (1030..1087 remain zero: action zero-pad to 50 + K pad)
    }
    for (int i = tid; i < 16 * EE; i += NTHREADS) {
      int r = i / EE, e = i % EE;
      pin[r * 1760 + 200 + e] = (__bf16)p.embed[((size_t)(b0 + r) * TT + t) * EE + e];
    }
    __syncthreads();

    // ---- img_in: [16,1088] @ W_in -> 200, LN, elu ----
    mm_wmma(x_in, 1088, p.ws + OFF_IN, KT_IN, NT_IN, fbuf, 1024);
    __syncthreads();
    ln_stats_wave(fbuf + wave * 1024, p.b_in, HID, &mu[wave], &rs[wave]);
    __syncthreads();
    for (int i = tid; i < 16 * HID; i += NTHREADS) {
      int r = i / HID, j = i % HID;
      float v = (fbuf[r * 1024 + j] + p.b_in[j] - mu[r]) * rs[r] * p.g_in[j] + p.be_in[j];
      v = v > 0.f ? v : expm1f(v);
      gru_in[r * 416 + j] = (__bf16)v;
      gru_in[r * 416 + 200 + j] = (__bf16)det[r * 200 + j];
    }
    __syncthreads();

    // ---- GRU: [16,416] @ W_gru -> 600, LN, gates ----
    mm_wmma(gru_in, 416, p.ws + OFF_GRU, KT_GRU, NT_GRU, fbuf, 1024);
    __syncthreads();
    ln_stats_wave(fbuf + wave * 1024, p.b_gru, 600, &mu[wave], &rs[wave]);
    __syncthreads();
    for (int i = tid; i < 16 * DET; i += NTHREADS) {
      int r = i / DET, j = i % DET;
      float m_ = mu[r], s_ = rs[r];
      float pr = (fbuf[r * 1024 + j]       + p.b_gru[j]       - m_) * s_ * p.g_gru[j]       + p.be_gru[j];
      float pc = (fbuf[r * 1024 + 200 + j] + p.b_gru[200 + j] - m_) * s_ * p.g_gru[200 + j] + p.be_gru[200 + j];
      float pu = (fbuf[r * 1024 + 400 + j] + p.b_gru[400 + j] - m_) * s_ * p.g_gru[400 + j] + p.be_gru[400 + j];
      float rg = 1.f / (1.f + expf(-pr));
      float cg = tanhf(rg * pc);
      float ug = 1.f / (1.f + expf(-(pu - 1.f)));
      float d  = ug * cg + (1.f - ug) * det[r * 200 + j];
      det[r * 200 + j] = d;
      __bf16 db = (__bf16)d;
      dbf[r * 224 + j] = db;
      pin[r * 1760 + j] = db;
      p.out_det[((size_t)(b0 + r) * TT + t) * DET + j] = d;
    }
    __syncthreads();

    const int eix = p.ens_idx[t];

    // ---- ensemble prior (selected head only; outputs identical) ----
    mm_wmma(dbf, 224, p.ws + OFF_1E + (size_t)eix * SZ_1E, KT_H, NT_HID, fbuf, 1024);
    __syncthreads();
    ln_stats_wave(fbuf + wave * 1024, p.b1e + (size_t)eix * HID, HID, &mu[wave], &rs[wave]);
    __syncthreads();
    for (int i = tid; i < 16 * HID; i += NTHREADS) {
      int r = i / HID, j = i % HID;
      const size_t o = (size_t)eix * HID + j;
      float v = (fbuf[r * 1024 + j] + p.b1e[o] - mu[r]) * rs[r] * p.g1e[o] + p.be1e[o];
      v = v > 0.f ? v : expm1f(v);
      hid[r * 224 + j] = (__bf16)v;
    }
    __syncthreads();
    mm_wmma(hid, 224, p.ws + OFF_2E + (size_t)eix * SZ_2E, KT_H, NT_SD, fbuf, 1024);
    __syncthreads();
    for (int i = tid; i < 16 * SD; i += NTHREADS) {
      int r = i >> 10, j = i & 1023;
      p.out_prl[((size_t)(b0 + r) * TT + t) * SD + j] =
          fbuf[r * 1024 + j] + p.b2e[(size_t)eix * SD + j];
    }
    {   // 16 rows x 32 dists = 512 argmax tasks; one per thread
      int r = tid >> 5, d = tid & 31;
      const float* fr = fbuf + r * 1024 + d * 32;
      const float* br = p.b2e + (size_t)eix * SD + d * 32;
      float best = fr[0] + br[0]; int bi = 0;
      for (int jj = 1; jj < 32; ++jj) {
        float v = fr[jj] + br[jj];
        if (v > best) { best = v; bi = jj; }
      }
      size_t ob = ((size_t)(b0 + r) * TT + t) * SD + d * 32;
      for (int jj = 0; jj < 32; ++jj) p.out_prs[ob + jj] = (jj == bi) ? 1.f : 0.f;
    }
    __syncthreads();

    // ---- posterior: [16,1760] @ Wo1 -> 200, LN, elu; then @ Wo2 -> 1024 ----
    mm_wmma(pin, 1760, p.ws + OFF_O1, KT_O1, NT_HID, fbuf, 1024);
    __syncthreads();
    ln_stats_wave(fbuf + wave * 1024, p.bo1, HID, &mu[wave], &rs[wave]);
    __syncthreads();
    for (int i = tid; i < 16 * HID; i += NTHREADS) {
      int r = i / HID, j = i % HID;
      float v = (fbuf[r * 1024 + j] + p.bo1[j] - mu[r]) * rs[r] * p.go1[j] + p.beo1[j];
      v = v > 0.f ? v : expm1f(v);
      hid[r * 224 + j] = (__bf16)v;
    }
    __syncthreads();
    mm_wmma(hid, 224, p.ws + OFF_O2, KT_H, NT_SD, fbuf, 1024);
    __syncthreads();
    for (int i = tid; i < 16 * SD; i += NTHREADS) {
      int r = i >> 10, j = i & 1023;
      p.out_pq[((size_t)(b0 + r) * TT + t) * SD + j] = fbuf[r * 1024 + j] + p.bo2[j];
    }
    {
      int r = tid >> 5, d = tid & 31;
      const float* fr = fbuf + r * 1024 + d * 32;
      const float* br = p.bo2 + d * 32;
      float best = fr[0] + br[0]; int bi = 0;
      for (int jj = 1; jj < 32; ++jj) {
        float v = fr[jj] + br[jj];
        if (v > best) { best = v; bi = jj; }
      }
      size_t ob = ((size_t)(b0 + r) * TT + t) * SD + d * 32;
      for (int jj = 0; jj < 32; ++jj) {
        float o = (jj == bi) ? 1.f : 0.f;
        p.out_ps[ob + jj] = o;                    // post_stoch output
        x_in[r * 1088 + d * 32 + jj] = (__bf16)o; // feeds next step
      }
    }
    __syncthreads();
  }
}

// ---------------------------------------------------------------------------
extern "C" void kernel_launch(void* const* d_in, const int* in_sizes, int n_in,
                              void* d_out, int out_size, void* d_ws, size_t ws_size,
                              hipStream_t stream) {
  (void)in_sizes; (void)n_in; (void)out_size; (void)ws_size;

  const float* embed    = (const float*)d_in[0];
  const float* action   = (const float*)d_in[1];
  const unsigned char* is_first = (const unsigned char*)d_in[2];
  const int*   ens_idx  = (const int*)d_in[3];
  const float* W_in  = (const float*)d_in[4];
  const float* b_in  = (const float*)d_in[5];
  const float* g_in  = (const float*)d_in[6];
  const float* be_in = (const float*)d_in[7];
  const float* W_gru  = (const float*)d_in[8];
  const float* b_gru  = (const float*)d_in[9];
  const float* g_gru  = (const float*)d_in[10];
  const float* be_gru = (const float*)d_in[11];
  const float* W1e  = (const float*)d_in[12];
  const float* b1e  = (const float*)d_in[13];
  const float* g1e  = (const float*)d_in[14];
  const float* be1e = (const float*)d_in[15];
  const float* W2e  = (const float*)d_in[16];
  const float* b2e  = (const float*)d_in[17];
  const float* Wo1  = (const float*)d_in[18];
  const float* bo1  = (const float*)d_in[19];
  const float* go1  = (const float*)d_in[20];
  const float* beo1 = (const float*)d_in[21];
  const float* Wo2  = (const float*)d_in[22];
  const float* bo2  = (const float*)d_in[23];

  __bf16* ws = (__bf16*)d_ws;

  auto conv = [&](const float* src, size_t off, int K, int N, int nkt, int nnt) {
    int total = nkt * nnt * TILE_ELEMS;
    rssm_convert_tiles<<<(total + 255) / 256, 256, 0, stream>>>(src, ws + off, K, N, nkt, nnt);
  };
  conv(W_in,  OFF_IN,  INDIM, HID,  KT_IN,  NT_IN);
  conv(W_gru, OFF_GRU, 400,   600,  KT_GRU, NT_GRU);
  for (int k = 0; k < 5; ++k)
    conv(W1e + (size_t)k * 200 * 200,  OFF_1E + (size_t)k * SZ_1E, 200, 200,  KT_H, NT_HID);
  for (int k = 0; k < 5; ++k)
    conv(W2e + (size_t)k * 200 * 1024, OFF_2E + (size_t)k * SZ_2E, 200, 1024, KT_H, NT_SD);
  conv(Wo1, OFF_O1, 1736, 200,  KT_O1, NT_HID);
  conv(Wo2, OFF_O2, 200,  1024, KT_H,  NT_SD);

  float* out = (float*)d_out;
  const size_t sz1 = (size_t)BB * TT * SD;        // 16,777,216
  const size_t szd = (size_t)BB * TT * DET;       //  3,276,800

  RssmParams p;
  p.embed = embed; p.action = action; p.is_first = is_first; p.ens_idx = ens_idx;
  p.b_in = b_in; p.g_in = g_in; p.be_in = be_in;
  p.b_gru = b_gru; p.g_gru = g_gru; p.be_gru = be_gru;
  p.b1e = b1e; p.g1e = g1e; p.be1e = be1e; p.b2e = b2e;
  p.bo1 = bo1; p.go1 = go1; p.beo1 = beo1; p.bo2 = bo2;
  p.ws = ws;
  p.out_ps  = out;
  p.out_pq  = out + sz1;
  p.out_det = out + 2 * sz1;
  p.out_prs = out + 2 * sz1 + szd;
  p.out_prl = out + 2 * sz1 + szd + sz1;

  rssm_scan_kernel<<<dim3(4), dim3(NTHREADS), SMEM_BYTES, stream>>>(p);
}
